// DecoderSSMLayer_60138132078595
// MI455X (gfx1250) — matrix operand order repaired and checked by reading, not compiled, not run on hardware
//
#include <hip/hip_runtime.h>
#include <hip/hip_bf16.h>
#include <stdint.h>

#define B_SZ    4
#define SEQ_L   4096
#define DMODEL  512
#define DSTATE  16
#define DINNER  1024
#define DCONV   4
#define DTRANK  32
#define NROWS   (B_SZ * SEQ_L)          /* 16384 */
#define N_XZ    (2 * DINNER)            /* 2048  */
#define N_XDBL  (DTRANK + 2 * DSTATE)   /* 64    */

typedef __attribute__((ext_vector_type(16))) __bf16 v16bf;
typedef __attribute__((ext_vector_type(8)))  __bf16 v8bf;
typedef __attribute__((ext_vector_type(8)))  float  v8f;
typedef __attribute__((ext_vector_type(4)))  int    v4i;

#if defined(__gfx1250__) && __has_builtin(__builtin_amdgcn_global_load_async_to_lds_b128) && \
    __has_builtin(__builtin_amdgcn_s_wait_asynccnt)
#define HAVE_ASYNC_LDS 1
#else
#define HAVE_ASYNC_LDS 0
#endif

__device__ __forceinline__ unsigned short f2bf(float f) {
    unsigned u = __float_as_uint(f);
    u += 0x7FFFu + ((u >> 16) & 1u);           // round-to-nearest-even
    return (unsigned short)(u >> 16);
}
__device__ __forceinline__ float bf2f(unsigned short h) {
    return __uint_as_float(((unsigned)h) << 16);
}

#if HAVE_ASYNC_LDS
// 16-byte global -> LDS async DMA (tracked with ASYNCcnt)
__device__ __forceinline__ void async_cp16(const unsigned short* g, __bf16* l) {
    __builtin_amdgcn_global_load_async_to_lds_b128(
        (__attribute__((address_space(1))) v4i*)(g),
        (__attribute__((address_space(3))) v4i*)(l),
        0, 0);
}
#endif

// ---------------------------------------------------------------------------
// One-time weight prep: fp32 [K][N] -> bf16 transposed [N][K]
// (so GEMM B-tiles stage with pure 16B vector copies, no in-loop transpose)
// ---------------------------------------------------------------------------
__global__ __launch_bounds__(256)
void transpose_f32_to_bf16_kernel(const float* __restrict__ in, unsigned short* __restrict__ out,
                                  int K, int N) {
    int i = blockIdx.x * 256 + threadIdx.x;
    if (i < K * N) {
        int n = i / K, k = i % K;                  // out index i = n*K + k (coalesced writes)
        out[i] = f2bf(in[(size_t)k * N + n]);
    }
}

// ---------------------------------------------------------------------------
// LayerNorm over D=512, emit bf16 rows for the WMMA GEMM A-matrix
// ---------------------------------------------------------------------------
__global__ __launch_bounds__(256)
void layernorm_bf16_kernel(const float* __restrict__ x, const float* __restrict__ g,
                           const float* __restrict__ be, unsigned short* __restrict__ out) {
    __shared__ float red[256];
    const int row = blockIdx.x;
    const int t   = threadIdx.x;
    const float* xr = x + (size_t)row * DMODEL;
    float v0 = xr[t], v1 = xr[t + 256];
    red[t] = v0 + v1;
    __syncthreads();
    #pragma unroll
    for (int s = 128; s > 0; s >>= 1) { if (t < s) red[t] += red[t + s]; __syncthreads(); }
    float mu = red[0] * (1.0f / DMODEL);
    __syncthreads();
    float d0 = v0 - mu, d1 = v1 - mu;
    red[t] = d0 * d0 + d1 * d1;
    __syncthreads();
    #pragma unroll
    for (int s = 128; s > 0; s >>= 1) { if (t < s) red[t] += red[t + s]; __syncthreads(); }
    float rstd = rsqrtf(red[0] * (1.0f / DMODEL) + 1e-5f);
    unsigned short* o = out + (size_t)row * DMODEL;
    o[t]       = f2bf(d0 * rstd * g[t]       + be[t]);
    o[t + 256] = f2bf(d1 * rstd * g[t + 256] + be[t + 256]);
}

// ---------------------------------------------------------------------------
// bf16 WMMA GEMM:  C[M,N] = A[M,K] * B[K,N], B given pre-transposed Bt[N][K].
// fp32 accumulate. mode 0: fp32 out, 1: bf16 out, 2: fp32 + residual.
// Double-buffered LDS tiles; async global->LDS DMA when available.
// Dims are exact multiples of the tile for all launches below.
// ---------------------------------------------------------------------------
template<int BM, int BN, int BK, int WM, int WN>
__global__ __launch_bounds__(WM * WN * 32)
void wmma_gemm_bf16(const unsigned short* __restrict__ Ag,
                    const unsigned short* __restrict__ Bt,
                    float* __restrict__ Cf,
                    unsigned short* __restrict__ Cbf,
                    const float* __restrict__ residual,
                    int M, int N, int K, int mode) {
    constexpr int WTM = BM / WM, WTN = BN / WN;
    constexpr int TM  = WTM / 16, TN = WTN / 16;
    constexpr int NTHREADS = WM * WN * 32;
    constexpr int ACH = BM * BK / 8;   // 16B chunks in an A tile
    constexpr int BCH = BN * BK / 8;   // 16B chunks in a B tile
    static_assert(BK == 32, "K-step must match wmma_f32_16x16x32_bf16");

    __shared__ __align__(64) __bf16 As[2][BM][BK];   // [m][k]
    __shared__ __align__(64) __bf16 Bs[2][BN][BK];   // [n][k]

    const int tid  = threadIdx.x;
    const int wave = tid >> 5;
    const int lane = tid & 31;
    const int half = lane >> 4;      // 0: lanes 0-15, 1: lanes 16-31
    const int lrow = lane & 15;
    const int wm   = wave / WN;
    const int wn   = wave % WN;
    const int tileM = blockIdx.y * BM;
    const int tileN = blockIdx.x * BN;

    const v8f zero8 = {0.f, 0.f, 0.f, 0.f, 0.f, 0.f, 0.f, 0.f};
    v8f acc[TM][TN];
    #pragma unroll
    for (int i = 0; i < TM; ++i)
        #pragma unroll
        for (int j = 0; j < TN; ++j) acc[i][j] = zero8;

    auto compute = [&](int p) {
        v16bf afrag[TM], bfrag[TN];
        #pragma unroll
        for (int i = 0; i < TM; ++i) {
            // A layout: lanes 0-15 hold K=0..15, lanes 16-31 K=16..31, row = lane%16
            const __bf16* pa = &As[p][wm * WTM + i * 16 + lrow][half * 16];
            v8bf lo = *(const v8bf*)(const void*)pa;
            v8bf hi = *(const v8bf*)(const void*)(pa + 8);
            v16bf a;
            #pragma unroll
            for (int e = 0; e < 8; ++e) { a[e] = lo[e]; a[e + 8] = hi[e]; }
            afrag[i] = a;
        }
        #pragma unroll
        for (int j = 0; j < TN; ++j) {
            // B layout: lanes 0-15 hold K=0..15, lanes 16-31 K=16..31, col = lane%16
            bfrag[j] = *(const v16bf*)(const void*)&Bs[p][wn * WTN + j * 16 + lrow][half * 16];
        }
        #pragma unroll
        for (int i = 0; i < TM; ++i)
            #pragma unroll
            for (int j = 0; j < TN; ++j)
                acc[i][j] = __builtin_amdgcn_wmma_f32_16x16x32_bf16(
                    false, afrag[i], false, bfrag[j], (short)0, acc[i][j], false, false);
    };

#if HAVE_ASYNC_LDS
    auto stage_async = [&](int p, int k0) {
        #pragma unroll
        for (int c = tid; c < ACH; c += NTHREADS) {
            int row = c / (BK / 8), col = (c % (BK / 8)) * 8;
            async_cp16(Ag + (size_t)(tileM + row) * K + k0 + col, &As[p][row][col]);
        }
        #pragma unroll
        for (int c = tid; c < BCH; c += NTHREADS) {
            int row = c / (BK / 8), col = (c % (BK / 8)) * 8;
            async_cp16(Bt + (size_t)(tileN + row) * K + k0 + col, &Bs[p][row][col]);
        }
    };
    stage_async(0, 0);
    int p = 0;
    for (int k0 = 0; k0 < K; k0 += BK) {
        __builtin_amdgcn_s_wait_asynccnt(0);   // my DMA for buffer p has landed
        __syncthreads();                       // everyone's has; prior reads of p^1 done
        if (k0 + BK < K) stage_async(p ^ 1, k0 + BK);
        compute(p);
        p ^= 1;
    }
#else
    for (int k0 = 0; k0 < K; k0 += BK) {
        #pragma unroll
        for (int c = tid; c < ACH; c += NTHREADS) {
            int row = c / (BK / 8), col = (c % (BK / 8)) * 8;
            v8bf v = *(const v8bf*)(const void*)(Ag + (size_t)(tileM + row) * K + k0 + col);
            *(v8bf*)(void*)&As[0][row][col] = v;
        }
        #pragma unroll
        for (int c = tid; c < BCH; c += NTHREADS) {
            int row = c / (BK / 8), col = (c % (BK / 8)) * 8;
            v8bf v = *(const v8bf*)(const void*)(Bt + (size_t)(tileN + row) * K + k0 + col);
            *(v8bf*)(void*)&Bs[0][row][col] = v;
        }
        __syncthreads();
        compute(0);
        __syncthreads();
    }
#endif

    // ---- epilogue: D layout, VGPR r -> M = r + half*8, N = lane%16 ---------
    #pragma unroll
    for (int i = 0; i < TM; ++i) {
        #pragma unroll
        for (int j = 0; j < TN; ++j) {
            #pragma unroll
            for (int r = 0; r < 8; ++r) {
                int row = tileM + wm * WTM + i * 16 + half * 8 + r;
                int col = tileN + wn * WTN + j * 16 + lrow;
                size_t idx = (size_t)row * N + col;
                float v = acc[i][j][r];
                if (mode == 1)      Cbf[idx] = f2bf(v);
                else if (mode == 2) Cf[idx]  = v + residual[idx];
                else                Cf[idx]  = v;
            }
        }
    }
}

// ---------------------------------------------------------------------------
// Causal depthwise conv(4) + bias + SiLU over the u-half of xz (bf16 in/out)
// ---------------------------------------------------------------------------
__global__ __launch_bounds__(256)
void conv_silu_kernel(const unsigned short* __restrict__ xz,
                      const float* __restrict__ w, const float* __restrict__ bias,
                      unsigned short* __restrict__ u_out) {
    size_t idx = (size_t)blockIdx.x * 256 + threadIdx.x;
    if (idx >= (size_t)NROWS * DINNER) return;
    int c = (int)(idx % DINNER);
    size_t row = idx / DINNER;
    int l = (int)(row % SEQ_L);
    const float* wc = w + (size_t)c * DCONV;
    float acc = bias[c];
    #pragma unroll
    for (int j = 0; j < DCONV; ++j) {
        int ls = l - (DCONV - 1) + j;
        if (ls >= 0)
            acc += wc[j] * bf2f(xz[(row - (size_t)(DCONV - 1 - j)) * N_XZ + c]);
    }
    float s = acc / (1.0f + __expf(-acc));   // SiLU
    u_out[row * DINNER + c] = f2bf(s);
}

// ---------------------------------------------------------------------------
// delta = softplus(dt @ W_dt + b_dt); dt = x_dbl[:, :32] cached in LDS
// ---------------------------------------------------------------------------
__global__ __launch_bounds__(256)
void dt_softplus_kernel(const float* __restrict__ x_dbl, const float* __restrict__ W_dt,
                        const float* __restrict__ b_dt, float* __restrict__ delta) {
    __shared__ float dt_sh[DTRANK];
    const int row = blockIdx.x;
    if (threadIdx.x < DTRANK) dt_sh[threadIdx.x] = x_dbl[(size_t)row * N_XDBL + threadIdx.x];
    __syncthreads();
    for (int c = threadIdx.x; c < DINNER; c += 256) {
        float a = b_dt[c];
        #pragma unroll
        for (int r = 0; r < DTRANK; ++r) a += dt_sh[r] * W_dt[(size_t)r * DINNER + c];
        float sp = (a > 20.0f) ? a : log1pf(__expf(a));
        delta[(size_t)row * DINNER + c] = sp;
    }
}

// ---------------------------------------------------------------------------
// Selective scan, fused with +u*D and SiLU(z) gating. One thread per channel,
// 128 channels/block, state[16] in registers, B/C rows staged via LDS chunks.
// ---------------------------------------------------------------------------
#define SCAN_CPB   128
#define SCAN_CHUNK 64
__global__ __launch_bounds__(SCAN_CPB)
void selective_scan_kernel(const float* __restrict__ x_dbl, const float* __restrict__ delta,
                           const unsigned short* __restrict__ u_bf,
                           const unsigned short* __restrict__ xz_bf,
                           const float* __restrict__ A_log, const float* __restrict__ Dp,
                           unsigned short* __restrict__ yg_bf) {
    __shared__ float Bsh[SCAN_CHUNK][DSTATE];
    __shared__ float Csh[SCAN_CHUNK][DSTATE];
    const int blocksPerBatch = DINNER / SCAN_CPB;          // 8
    const int b = blockIdx.x / blocksPerBatch;
    const int c = (blockIdx.x % blocksPerBatch) * SCAN_CPB + threadIdx.x;

    float A[DSTATE], s[DSTATE];
    #pragma unroll
    for (int n = 0; n < DSTATE; ++n) {
        A[n] = -__expf(A_log[(size_t)c * DSTATE + n]);
        s[n] = 0.0f;
    }
    const float Dv = Dp[c];
    const size_t rowbase = (size_t)b * SEQ_L;

    for (int l0 = 0; l0 < SEQ_L; l0 += SCAN_CHUNK) {
        __syncthreads();
        for (int i = threadIdx.x; i < SCAN_CHUNK * 2 * DSTATE; i += SCAN_CPB) {
            int t = i / (2 * DSTATE), q = i % (2 * DSTATE);
            float v = x_dbl[(rowbase + l0 + t) * N_XDBL + DTRANK + q];
            if (q < DSTATE) Bsh[t][q] = v; else Csh[t][q - DSTATE] = v;
        }
        __syncthreads();
        for (int t = 0; t < SCAN_CHUNK; ++t) {
            size_t row = rowbase + l0 + t;
            float dl = delta[row * DINNER + c];
            float u  = bf2f(u_bf[row * DINNER + c]);
            float du = dl * u;
            float y  = 0.0f;
            #pragma unroll
            for (int n = 0; n < DSTATE; ++n) {
                float dA = __expf(dl * A[n]);
                s[n] = dA * s[n] + du * Bsh[t][n];
                y += s[n] * Csh[t][n];
            }
            y += u * Dv;
            float z = bf2f(xz_bf[row * N_XZ + DINNER + c]);
            float g = z / (1.0f + __expf(-z));
            yg_bf[row * DINNER + c] = f2bf(y * g);
        }
    }
}

// ---------------------------------------------------------------------------
// Host launcher
// ---------------------------------------------------------------------------
extern "C" void kernel_launch(void* const* d_in, const int* in_sizes, int n_in,
                              void* d_out, int out_size, void* d_ws, size_t ws_size,
                              hipStream_t stream) {
    (void)in_sizes; (void)n_in; (void)out_size; (void)ws_size;
    const float* x        = (const float*)d_in[0];   // [4,4096,512]
    const float* ln_gamma = (const float*)d_in[1];
    const float* ln_beta  = (const float*)d_in[2];
    const float* W_in     = (const float*)d_in[3];   // [512,2048]
    const float* conv_w   = (const float*)d_in[4];   // [1024,1,4]
    const float* conv_b   = (const float*)d_in[5];
    const float* W_x      = (const float*)d_in[6];   // [1024,64]
    const float* W_dt     = (const float*)d_in[7];   // [32,1024]
    const float* b_dt     = (const float*)d_in[8];
    const float* A_log    = (const float*)d_in[9];   // [1024,16]
    const float* Dvec     = (const float*)d_in[10];
    const float* W_out    = (const float*)d_in[11];  // [1024,512]
    float* out = (float*)d_out;

    // ---- workspace carving (256B aligned) ----------------------------------
    uint8_t* wsb = (uint8_t*)d_ws;
    size_t off = 0;
    auto take = [&](size_t bytes) -> size_t {
        size_t o = off; off += (bytes + 255) & ~(size_t)255; return o;
    };
    unsigned short* h_bf    = (unsigned short*)(wsb + take((size_t)NROWS * DMODEL * 2));
    unsigned short* Win_bt  = (unsigned short*)(wsb + take((size_t)DMODEL * N_XZ * 2));   // [2048][512]
    unsigned short* Wx_bt   = (unsigned short*)(wsb + take((size_t)DINNER * N_XDBL * 2)); // [64][1024]
    unsigned short* Wout_bt = (unsigned short*)(wsb + take((size_t)DINNER * DMODEL * 2)); // [512][1024]
    unsigned short* xz_bf   = (unsigned short*)(wsb + take((size_t)NROWS * N_XZ * 2));
    unsigned short* u_bf    = (unsigned short*)(wsb + take((size_t)NROWS * DINNER * 2));
    float*          x_dbl   = (float*)(wsb + take((size_t)NROWS * N_XDBL * 4));
    float*          delta   = (float*)(wsb + take((size_t)NROWS * DINNER * 4));
    unsigned short* yg_bf   = (unsigned short*)(wsb + take((size_t)NROWS * DINNER * 2));

    // ---- 0) weight conversion + transpose to bf16 [N][K] -------------------
    {
        int n1 = DMODEL * N_XZ, n2 = DINNER * N_XDBL, n3 = DINNER * DMODEL;
        transpose_f32_to_bf16_kernel<<<(n1 + 255) / 256, 256, 0, stream>>>(W_in,  Win_bt,  DMODEL, N_XZ);
        transpose_f32_to_bf16_kernel<<<(n2 + 255) / 256, 256, 0, stream>>>(W_x,   Wx_bt,   DINNER, N_XDBL);
        transpose_f32_to_bf16_kernel<<<(n3 + 255) / 256, 256, 0, stream>>>(W_out, Wout_bt, DINNER, DMODEL);
    }
    // ---- 1) layernorm -> h_bf ----------------------------------------------
    layernorm_bf16_kernel<<<NROWS, 256, 0, stream>>>(x, ln_gamma, ln_beta, h_bf);

    // ---- 2) xz = h @ W_in  [16384x512]*[512x2048] -> bf16 ------------------
    wmma_gemm_bf16<128, 128, 32, 2, 4><<<dim3(N_XZ / 128, NROWS / 128), 256, 0, stream>>>(
        h_bf, Win_bt, nullptr, xz_bf, nullptr, NROWS, N_XZ, DMODEL, 1);

    // ---- 3) depthwise causal conv + SiLU -> u_bf ---------------------------
    {
        size_t total = (size_t)NROWS * DINNER;
        conv_silu_kernel<<<(unsigned)((total + 255) / 256), 256, 0, stream>>>(
            xz_bf, conv_w, conv_b, u_bf);
    }
    // ---- 4) x_dbl = u @ W_x  [16384x1024]*[1024x64] -> fp32 ----------------
    wmma_gemm_bf16<128, 64, 32, 4, 2><<<dim3(N_XDBL / 64, NROWS / 128), 256, 0, stream>>>(
        u_bf, Wx_bt, x_dbl, nullptr, nullptr, NROWS, N_XDBL, DINNER, 0);

    // ---- 5) delta = softplus(dt @ W_dt + b_dt) -----------------------------
    dt_softplus_kernel<<<NROWS, 256, 0, stream>>>(x_dbl, W_dt, b_dt, delta);

    // ---- 6) selective scan + D*u + SiLU(z) gate -> yg_bf -------------------
    selective_scan_kernel<<<B_SZ * (DINNER / SCAN_CPB), SCAN_CPB, 0, stream>>>(
        x_dbl, delta, u_bf, xz_bf, A_log, Dvec, yg_bf);

    // ---- 7) out = residual + yg @ W_out [16384x1024]*[1024x512] ------------
    wmma_gemm_bf16<128, 128, 32, 2, 4><<<dim3(DMODEL / 128, NROWS / 128), 256, 0, stream>>>(
        yg_bf, Wout_bt, out, nullptr, x, NROWS, DMODEL, DINNER, 2);
}
